// ParallelPalmBlock_23407571763843
// MI455X (gfx1250) — matrix operand order repaired and checked by reading, not compiled, not run on hardware
//
#include <hip/hip_runtime.h>

// ---------------------------------------------------------------------------
// ParallelPaLM block for MI455X (gfx1250, wave32, WMMA + TDM + async-to-LDS).
// Compute-bound: ~278 GFLOP vs ~237MB weights -> feed v_wmma_f32_16x16x32_bf16.
// ---------------------------------------------------------------------------

#define SEQ        2048
#define HIDDEN     2048
#define HEADS      16
#define DIM_HEAD   128
#define FF_INNER   8192
#define ATTN_INNER 2048
#define FUSED_DIM  (ATTN_INNER + DIM_HEAD + DIM_HEAD + FF_INNER + FF_INNER) // 18688
#define FF_OFF     (ATTN_INNER + 2*DIM_HEAD)            // 2304
#define GATE_OFF   (FF_OFF + FF_INNER)                  // 10496

typedef __attribute__((ext_vector_type(16))) __bf16          v16bf;
typedef __attribute__((ext_vector_type(8)))  float           v8f;
typedef __attribute__((ext_vector_type(8)))  unsigned short  us8;
typedef __attribute__((ext_vector_type(4)))  unsigned int    u32x4;
typedef __attribute__((ext_vector_type(8)))  unsigned int    u32x8;

union BF16Frag { v16bf bf; us8 h[2]; };

__device__ __forceinline__ unsigned short f2bf(float f) {
    union { float f; unsigned u; } c; c.f = f;
    unsigned r = c.u + 0x7FFFu + ((c.u >> 16) & 1u);   // round-to-nearest-even
    return (unsigned short)(r >> 16);
}
__device__ __forceinline__ float bf2f(unsigned short h) {
    union { unsigned u; float f; } c; c.u = ((unsigned)h) << 16;
    return c.f;
}

__device__ __forceinline__ v8f wmma_bf16(const BF16Frag& a, const BF16Frag& b, v8f c) {
    // D = A(16x32 bf16) * B(32x16 bf16) + C(16x16 f32)
    return __builtin_amdgcn_wmma_f32_16x16x32_bf16(
        false, a.bf, false, b.bf, (short)0, c, false, false);
}

// ---------------------------------------------------------------------------
// Tensor Data Mover: 2D bf16 tile, global -> LDS, with LDS row padding.
// D# groups 0-1 only (2D tensor; VADDR2/3 = NULL via short asm form).
// pad_interval_code 3 => pad after every 16 DWORDs (64B = one 32-elem row)
// pad_amount_code   3 => insert 4 DWORDs (16B) => 80B LDS row stride.
// ---------------------------------------------------------------------------
__device__ __forceinline__ void tdm_load_2d_bf16(
    unsigned lds_off, const void* gaddr,
    unsigned tile_w_elems, unsigned tile_h,
    unsigned tensor_w_elems, unsigned long long row_stride_elems,
    unsigned pad_interval_code, unsigned pad_amount_code)
{
    unsigned long long ga = (unsigned long long)(uintptr_t)gaddr;
    u32x4 g0;
    g0.x = 1u;                                            // count=1, user mode
    g0.y = lds_off;                                       // lds_addr (bytes)
    g0.z = (unsigned)(ga & 0xFFFFFFFFu);                  // global_addr[31:0]
    g0.w = (unsigned)((ga >> 32) & 0x01FFFFFFu)           // global_addr[56:32]
         | (2u << 30);                                    // type = 2 (image)
    u32x8 g1;
    g1.s0 = (1u << 16)                                    // data_size = 2 bytes
          | (1u << 20)                                    // pad_enable
          | (pad_interval_code << 22)
          | (pad_amount_code << 25);
    g1.s1 = (tensor_w_elems & 0xFFFFu) << 16;             // tensor_dim0[15:0]
    g1.s2 = ((tensor_w_elems >> 16) & 0xFFFFu)            // tensor_dim0[31:16]
          | (0xFFFFu << 16);                              // tensor_dim1[15:0]
    g1.s3 = 0x000Fu                                       // tensor_dim1[19:16] (big)
          | ((tile_w_elems & 0xFFFFu) << 16);             // tile_dim0
    g1.s4 = tile_h & 0xFFFFu;                             // tile_dim1 (tile_dim2=0)
    g1.s5 = (unsigned)(row_stride_elems & 0xFFFFFFFFu);   // tensor_dim0_stride[31:0]
    g1.s6 = (unsigned)((row_stride_elems >> 32) & 0xFFFFu); // stride[47:32]
    g1.s7 = 0u;                                           // tensor_dim1_stride = 0 (2D)
    asm volatile("tensor_load_to_lds %0, %1" :: "s"(g0), "s"(g1) : "memory");
}
__device__ __forceinline__ void wait_tensor0() {
    asm volatile("s_wait_tensorcnt 0x0" ::: "memory");
}

// ---------------------------------------------------------------------------
// Per-lane async global -> LDS copy (ASYNCcnt tracked).
// ---------------------------------------------------------------------------
__device__ __forceinline__ void async_copy_b128(void* lds, const void* g) {
    unsigned l = (unsigned)(uintptr_t)lds;                // low 32 bits = LDS offset
    unsigned long long ga = (unsigned long long)(uintptr_t)g;
    asm volatile("global_load_async_to_lds_b128 %0, %1, off"
                 :: "v"(l), "v"(ga) : "memory");
}
__device__ __forceinline__ void wait_async0() {
    asm volatile("s_wait_asynccnt 0x0" ::: "memory");
}

// ---------------------------------------------------------------------------
// RMSNorm: one block per row of [SEQ, HIDDEN]; output bf16.
// ---------------------------------------------------------------------------
__global__ __launch_bounds__(256) void rmsnorm_bf16_kernel(
    const float* __restrict__ x, const float* __restrict__ w,
    unsigned short* __restrict__ xn)
{
    int row = blockIdx.x;
    int tid = threadIdx.x;
    const float* xr = x + (size_t)row * HIDDEN;
    float v[8];
    float ss = 0.f;
#pragma unroll
    for (int i = 0; i < 8; ++i) { v[i] = xr[tid + i * 256]; ss += v[i] * v[i]; }
#pragma unroll
    for (int m = 1; m < 32; m <<= 1) ss += __shfl_xor(ss, m, 32);
    __shared__ float red[8];
    if ((tid & 31) == 0) red[tid >> 5] = ss;
    __syncthreads();
    float tot = 0.f;
#pragma unroll
    for (int i = 0; i < 8; ++i) tot += red[i];
    float inv = rsqrtf(tot / (float)HIDDEN + 1e-6f);
#pragma unroll
    for (int i = 0; i < 8; ++i) {
        int c = tid + i * 256;
        xn[(size_t)row * HIDDEN + c] = f2bf(v[i] * inv * w[c]);
    }
}

// ---------------------------------------------------------------------------
// WMMA GEMM: C[M,N] = A_bf16[M,K] * B_f32->bf16[K,N].
// Block tile 128x64, 8 waves, wave tile 32x32 (2x2 wmma), K-step 32.
// A tile staged by the Tensor Data Mover (wave 0 issues one 2D descriptor,
// TDM pads LDS rows 64B->80B). B tile: load fp32 -> cvt bf16 -> LDS transposed.
// EPI: 0 = store bf16, 1 = store f32, 2 = accumulate f32.
// ---------------------------------------------------------------------------
template <int EPI>
__global__ __launch_bounds__(256) void gemm_bf16w_kernel(
    const unsigned short* __restrict__ A, const float* __restrict__ B,
    void* __restrict__ Cv, int K, int lda, int ldb, int ldc)
{
    __shared__ unsigned short Asm[128][40];  // [m][k], 80B row stride (TDM-padded)
    __shared__ unsigned short Bsm[64][40];   // transposed: [n][k]
    int tid  = threadIdx.x;
    int wave = tid >> 5, lane = tid & 31;
    int half = lane >> 4, lr = lane & 15;
    int wm = (wave & 3) * 32, wn = (wave >> 2) * 32;
    int m0 = blockIdx.x * 128, n0 = blockIdx.y * 64;

    v8f acc[2][2] = {};

    for (int k0 = 0; k0 < K; k0 += 32) {
        // Kick off TDM copy of the A tile (128 rows x 32 bf16) into Asm.
        if (wave == 0) {
            tdm_load_2d_bf16((unsigned)(uintptr_t)&Asm[0][0],
                             A + (size_t)m0 * lda + k0,
                             /*tile_w=*/32, /*tile_h=*/128,
                             /*tensor_w=*/(unsigned)lda,
                             /*row_stride=*/(unsigned long long)lda,
                             /*pad_interval=*/3, /*pad_amount=*/3);
        }
        // Meanwhile: stage B tile 32x64 fp32 -> bf16, transposed into [n][k].
#pragma unroll
        for (int it = 0; it < 2; ++it) {
            int id = tid + it * 256;
            int row = id >> 4, seg = (id & 15) << 2;
            const float* bp = B + (size_t)(k0 + row) * ldb + n0 + seg;
            float4 bv = *(const float4*)bp;
            Bsm[seg + 0][row] = f2bf(bv.x);
            Bsm[seg + 1][row] = f2bf(bv.y);
            Bsm[seg + 2][row] = f2bf(bv.z);
            Bsm[seg + 3][row] = f2bf(bv.w);
        }
        if (k0 + 32 < K)
            __builtin_prefetch(B + (size_t)(k0 + 32 + (tid >> 3)) * ldb + n0, 0, 1);
        if (wave == 0) wait_tensor0();
        __syncthreads();

        // A frag: lane row = lr; vgprs 0-3 hold K=kg..kg+7, vgprs 4-7 K=kg+16..kg+23.
        BF16Frag a[2], b[2];
#pragma unroll
        for (int i = 0; i < 2; ++i) {
            int row = wm + i * 16 + lr, kg = half << 3;
            a[i].h[0] = *(const us8*)&Asm[row][kg];
            a[i].h[1] = *(const us8*)&Asm[row][16 + kg];
        }
        // B frag: lane col = lr; vgprs 0-3 hold K=ko..ko+7, vgprs 4-7 K=ko+16..ko+23.
#pragma unroll
        for (int j = 0; j < 2; ++j) {
            int col = wn + j * 16 + lr, ko = half << 3;
            b[j].h[0] = *(const us8*)&Bsm[col][ko];
            b[j].h[1] = *(const us8*)&Bsm[col][16 + ko];
        }
#pragma unroll
        for (int i = 0; i < 2; ++i)
#pragma unroll
            for (int j = 0; j < 2; ++j)
                acc[i][j] = wmma_bf16(a[i], b[j], acc[i][j]);
        __syncthreads();
    }

    // Epilogue: C layout = vgpr r -> M = r + 8*half, lane -> N = lr.
#pragma unroll
    for (int i = 0; i < 2; ++i)
#pragma unroll
        for (int j = 0; j < 2; ++j)
#pragma unroll
            for (int r = 0; r < 8; ++r) {
                int gm = m0 + wm + i * 16 + r + (half << 3);
                int gn = n0 + wn + j * 16 + lr;
                size_t idx = (size_t)gm * ldc + gn;
                if (EPI == 0)      ((unsigned short*)Cv)[idx] = f2bf(acc[i][j][r]);
                else if (EPI == 1) ((float*)Cv)[idx] = acc[i][j][r];
                else               ((float*)Cv)[idx] += acc[i][j][r];
            }
}

// ---------------------------------------------------------------------------
// RoPE in-place on q (per-head, scaled by 128^-0.5) and k regions of fused.
// ---------------------------------------------------------------------------
__global__ __launch_bounds__(256) void rope_kernel(
    const float* __restrict__ freqs, unsigned short* __restrict__ fused)
{
    const int PAIRS = HEADS * (DIM_HEAD / 2) + (DIM_HEAD / 2);  // 1088
    int idx = blockIdx.x * 256 + threadIdx.x;
    if (idx >= SEQ * PAIRS) return;
    int pos = idx / PAIRS;
    int r   = idx % PAIRS;
    size_t base;
    float scale;
    int j;
    if (r < HEADS * (DIM_HEAD / 2)) {
        int hh = r >> 6; j = r & 63;
        base  = (size_t)pos * FUSED_DIM + hh * DIM_HEAD + 2 * j;
        scale = 0.08838834764831845f;   // DIM_HEAD^-0.5
    } else {
        j = r - HEADS * (DIM_HEAD / 2);
        base  = (size_t)pos * FUSED_DIM + ATTN_INNER + 2 * j;
        scale = 1.f;
    }
    float ang = freqs[pos * (DIM_HEAD / 2) + j];
    float c = __cosf(ang), s = __sinf(ang);
    float xr = bf2f(fused[base]), xi = bf2f(fused[base + 1]);
    fused[base]     = f2bf((xr * c - xi * s) * scale);
    fused[base + 1] = f2bf((xr * s + xi * c) * scale);
}

// ---------------------------------------------------------------------------
// ffh = ff * swish(gate), bf16 in / bf16 out.
// ---------------------------------------------------------------------------
__global__ __launch_bounds__(256) void swishmul_kernel(
    const unsigned short* __restrict__ fused, unsigned short* __restrict__ ffh)
{
    size_t idx = (size_t)blockIdx.x * 256 + threadIdx.x;  // SEQ*FF_INNER, exact
    size_t row = idx / FF_INNER, col = idx % FF_INNER;
    size_t fb = row * FUSED_DIM;
    float f = bf2f(fused[fb + FF_OFF + col]);
    float g = bf2f(fused[fb + GATE_OFF + col]);
    float sw = g / (1.f + __expf(-g));
    ffh[idx] = f2bf(f * sw);
}

// ---------------------------------------------------------------------------
// Flash attention: block = (head, 128-query stripe), 8 waves x 16 queries.
// V tile arrives via global_load_async_to_lds_b128 (overlapped with QK^T),
// then an LDS->LDS transpose builds the WMMA B-layout tile.
// ---------------------------------------------------------------------------
__global__ __launch_bounds__(256) void attn_kernel(
    const unsigned short* __restrict__ fused, unsigned short* __restrict__ attnbuf)
{
    __shared__ unsigned short Vrm[32][128];    // V tile row-major (async dest)
    __shared__ unsigned short Vt[128][40];     // V tile transposed: [d][key]
    __shared__ unsigned short Pl[8][16][40];   // per-wave P staging [qrow][key]
    int h  = blockIdx.x, qb = blockIdx.y;
    int tid  = threadIdx.x;
    int wave = tid >> 5, lane = tid & 31;
    int half = lane >> 4, lr = lane & 15;
    int q0 = qb * 128 + wave * 16;
    int vkey  = tid >> 3;
    int vdseg = (tid & 7) << 4;

    // Q fragments: A-matrix 16x128 -> 4 chunks of K=32.
    BF16Frag qf[4];
    {
        const unsigned short* qp = fused + (size_t)(q0 + lr) * FUSED_DIM + h * DIM_HEAD;
        int kg = half << 3;
#pragma unroll
        for (int c = 0; c < 4; ++c) {
            qf[c].h[0] = *(const us8*)(qp + c * 32 + kg);
            qf[c].h[1] = *(const us8*)(qp + c * 32 + 16 + kg);
        }
    }

    v8f acc[8] = {};
    float Mrow[8], Lrow[8];
#pragma unroll
    for (int r = 0; r < 8; ++r) { Mrow[r] = -3.0e38f; Lrow[r] = 0.f; }

    int nsteps = (qb + 1) * 4;                 // keys [0, qb*128+128) in steps of 32
    for (int s = 0; s < nsteps; ++s) {
        int key0 = s * 32;

        // Kick off async V-tile copy (32 keys x 128 d, row-major) -> Vrm.
        {
            const unsigned short* vp = fused + (size_t)(key0 + vkey) * FUSED_DIM
                                     + ATTN_INNER + DIM_HEAD + vdseg;
            async_copy_b128(&Vrm[vkey][vdseg], vp);
            async_copy_b128(&Vrm[vkey][vdseg + 8], vp + 8);
        }

        // S = Q * K^T : two 16-key tiles, chained over d (4 x K=32).
        v8f s0 = {}, s1 = {};
#pragma unroll
        for (int c = 0; c < 4; ++c) {
            BF16Frag kf0, kf1;
            const unsigned short* kp = fused + (size_t)(key0 + lr) * FUSED_DIM
                                     + ATTN_INNER + c * 32 + (half << 3);
            kf0.h[0] = *(const us8*)kp;
            kf0.h[1] = *(const us8*)(kp + 16);
            const unsigned short* kp1 = kp + (size_t)16 * FUSED_DIM;
            kf1.h[0] = *(const us8*)kp1;
            kf1.h[1] = *(const us8*)(kp1 + 16);
            s0 = wmma_bf16(qf[c], kf0, s0);
            s1 = wmma_bf16(qf[c], kf1, s1);
        }
        // Online softmax update; row M = r + 8*half lives in 16 lanes of one half.
#pragma unroll
        for (int r = 0; r < 8; ++r) {
            int qv = q0 + r + (half << 3);
            bool ok0 = (key0 + lr) <= qv;
            bool ok1 = (key0 + 16 + lr) <= qv;
            float t = fmaxf(ok0 ? s0[r] : -3.0e38f, ok1 ? s1[r] : -3.0e38f);
#pragma unroll
            for (int mm = 1; mm < 16; mm <<= 1) t = fmaxf(t, __shfl_xor(t, mm, 32));
            float nm = fmaxf(Mrow[r], t);
            float sc = __expf(Mrow[r] - nm);
            Mrow[r] = nm;
            float p0 = ok0 ? __expf(s0[r] - nm) : 0.f;
            float p1 = ok1 ? __expf(s1[r] - nm) : 0.f;
            float rs = p0 + p1;
#pragma unroll
            for (int mm = 1; mm < 16; mm <<= 1) rs += __shfl_xor(rs, mm, 32);
            Lrow[r] = Lrow[r] * sc + rs;
#pragma unroll
            for (int j = 0; j < 8; ++j) acc[j][r] *= sc;
            int prow = r + (half << 3);
            Pl[wave][prow][lr]      = f2bf(p0);
            Pl[wave][prow][16 + lr] = f2bf(p1);
        }

        wait_async0();          // our wave's async V stores landed in LDS
        __syncthreads();        // Vrm + Pl published block-wide

        // Transpose Vrm -> Vt (WMMA B layout: [d][key]).
        {
            us8 v0 = *(const us8*)&Vrm[vkey][vdseg];
            us8 v1 = *(const us8*)&Vrm[vkey][vdseg + 8];
#pragma unroll
            for (int i = 0; i < 8; ++i) Vt[vdseg + i][vkey] = v0[i];
#pragma unroll
            for (int i = 0; i < 8; ++i) Vt[vdseg + 8 + i][vkey] = v1[i];
        }
        __syncthreads();

        // O += P(16x32) * V(32x128): 8 wmma over d tiles.
        BF16Frag pa;
        {
            int kg = half << 3;
            pa.h[0] = *(const us8*)&Pl[wave][lr][kg];
            pa.h[1] = *(const us8*)&Pl[wave][lr][16 + kg];
        }
#pragma unroll
        for (int j = 0; j < 8; ++j) {
            BF16Frag vf;
            int dcol = j * 16 + lr, ko = half << 3;
            vf.h[0] = *(const us8*)&Vt[dcol][ko];
            vf.h[1] = *(const us8*)&Vt[dcol][16 + ko];
            acc[j] = wmma_bf16(pa, vf, acc[j]);
        }
        __syncthreads();
    }

    // Normalize and write [seq][h*128+d] bf16.
#pragma unroll
    for (int j = 0; j < 8; ++j)
#pragma unroll
        for (int r = 0; r < 8; ++r) {
            int qv = q0 + r + (half << 3);
            attnbuf[(size_t)qv * ATTN_INNER + h * DIM_HEAD + j * 16 + lr] =
                f2bf(acc[j][r] / Lrow[r]);
        }
}

// ---------------------------------------------------------------------------
extern "C" void kernel_launch(void* const* d_in, const int* in_sizes, int n_in,
                              void* d_out, int out_size, void* d_ws, size_t ws_size,
                              hipStream_t stream) {
    (void)in_sizes; (void)n_in; (void)out_size; (void)ws_size;
    const float* hidden  = (const float*)d_in[0];
    const float* freqs   = (const float*)d_in[1];
    // d_in[2] = causal_mask (unused; mask computed analytically)
    const float* wi      = (const float*)d_in[3];
    const float* attn_wo = (const float*)d_in[4];
    const float* ff_wo   = (const float*)d_in[5];
    const float* norm_w  = (const float*)d_in[6];
    float* out = (float*)d_out;

    unsigned short* xn      = (unsigned short*)d_ws;                 // [SEQ][HIDDEN]
    unsigned short* fused   = xn + (size_t)SEQ * HIDDEN;             // [SEQ][FUSED_DIM]
    unsigned short* ffh     = fused + (size_t)SEQ * FUSED_DIM;       // [SEQ][FF_INNER]
    unsigned short* attnbuf = ffh + (size_t)SEQ * FF_INNER;          // [SEQ][ATTN_INNER]

    rmsnorm_bf16_kernel<<<SEQ, 256, 0, stream>>>(hidden, norm_w, xn);

    gemm_bf16w_kernel<0><<<dim3(SEQ / 128, FUSED_DIM / 64), 256, 0, stream>>>(
        xn, wi, fused, HIDDEN, HIDDEN, FUSED_DIM, FUSED_DIM);

    {
        int total = SEQ * (HEADS * (DIM_HEAD / 2) + DIM_HEAD / 2);
        rope_kernel<<<(total + 255) / 256, 256, 0, stream>>>(freqs, fused);
    }

    swishmul_kernel<<<(SEQ * (size_t)FF_INNER) / 256, 256, 0, stream>>>(fused, ffh);

    attn_kernel<<<dim3(HEADS, SEQ / 128), 256, 0, stream>>>(fused, attnbuf);

    gemm_bf16w_kernel<1><<<dim3(SEQ / 128, HIDDEN / 64), 256, 0, stream>>>(
        attnbuf, attn_wo, out, ATTN_INNER, ATTN_INNER, HIDDEN, HIDDEN);

    gemm_bf16w_kernel<2><<<dim3(SEQ / 128, HIDDEN / 64), 256, 0, stream>>>(
        ffh, ff_wo, out, FF_INNER, FF_INNER, HIDDEN, HIDDEN);
}
